// GraphPlanEncoder_7593502179348
// MI455X (gfx1250) — matrix operand order repaired and checked by reading, not compiled, not run on hardware
//
#include <hip/hip_runtime.h>
#include <math.h>

// ---------------- types ----------------
typedef __attribute__((ext_vector_type(16))) __bf16        bf16x16;
typedef __attribute__((ext_vector_type(8)))  float         f32x8;
typedef __attribute__((ext_vector_type(4)))  float         f32x4;
typedef __attribute__((ext_vector_type(4)))  unsigned int  u32x4;

#define N_NODES   100000
#define N_EDGES   600000
#define DIM       128
#define N_GRAPHS  256
#define OUT_DIM   256
#define MTILE     32          // rows per block (2 x 16-row WMMA tiles, B reused)

__device__ __forceinline__ unsigned short f32_to_bf16(float x) {
  unsigned int u = __float_as_uint(x);
  u += 0x7FFFu + ((u >> 16) & 1u);          // round to nearest even
  return (unsigned short)(u >> 16);
}

union AFrag {                                // 16 bf16 = 32 bytes = 8 VGPRs
  bf16x16 v;
  u32x4   q[2];
};

// ---------------- utility kernels ----------------
__global__ void fill_u32_kernel(unsigned int* p, unsigned int v, long long n) {
  long long i = (long long)blockIdx.x * blockDim.x + threadIdx.x;
  long long stride = (long long)gridDim.x * blockDim.x;
  for (; i < n; i += stride) p[i] = v;
}

__global__ void count_deg_kernel(const int* __restrict__ dst, float* __restrict__ cnt) {
  int e = blockIdx.x * blockDim.x + threadIdx.x;
  if (e < N_EDGES) atomicAdd(cnt + dst[e], 1.0f);
}

// W is [fin=128][fout=128] f32 row-major; write WT as bf16 [n=128][k=128]
__global__ void wtrans_bf16_kernel(const float* __restrict__ W, unsigned short* __restrict__ Wt) {
  int t = blockIdx.x * blockDim.x + threadIdx.x;      // 0..16383
  int n = t >> 7, k = t & 127;
  Wt[t] = f32_to_bf16(W[k * DIM + n]);
}

// ---------------- edge scatter (mean aggregation numerator) ----------------
__global__ void scatter_agg_kernel(const float* __restrict__ h,
                                   const int* __restrict__ src,
                                   const int* __restrict__ dst,
                                   float* __restrict__ agg) {
  long long t = (long long)blockIdx.x * blockDim.x + threadIdx.x;
  int e    = (int)(t >> 5);
  int lane = (int)(t & 31);
  if (e >= N_EDGES) return;
  int s = src[e], d = dst[e];
  f32x4 v = *(const f32x4*)(h + (long long)s * DIM + lane * 4);
  float* a = agg + (long long)d * DIM + lane * 4;
  atomicAdd(a + 0, v.x);
  atomicAdd(a + 1, v.y);
  atomicAdd(a + 2, v.z);
  atomicAdd(a + 3, v.w);
}

// ---------------- fused SAGE layer: mean@Wl + h@Wr + b -> LN -> ReLU ----------------
// 32 node rows per block; 8 waves; wave w owns output cols [16w,16w+16) for BOTH
// 16-row sub-tiles, so each B fragment load feeds two v_wmma ops.
__global__ __launch_bounds__(256)
void sage_wmma_kernel(const float* __restrict__ h_in,
                      const float* __restrict__ agg,
                      const float* __restrict__ cnt,
                      const unsigned short* __restrict__ WlT,   // bf16 [n][k]
                      const unsigned short* __restrict__ WrT,   // bf16 [n][k]
                      const float* __restrict__ bias,
                      const float* __restrict__ gamma,
                      const float* __restrict__ beta,
                      float* __restrict__ h_out) {
  __shared__ __align__(16) unsigned short sH[MTILE * DIM];   // h rows (bf16)   8KB
  __shared__ __align__(16) unsigned short sM[MTILE * DIM];   // mean rows (bf16)8KB
  __shared__ __align__(16) float          sO[MTILE * DIM];   // f32 tile out   16KB

  const int tid = threadIdx.x;
  const int m0  = blockIdx.x * MTILE;

  // warm L2/WGP$ for the 2x32KB bf16 weight matrices
  if (tid < 64) {
    __builtin_prefetch(WlT + tid * 256, 0, 0);
    __builtin_prefetch(WrT + tid * 256, 0, 0);
  }

  // ---- stage 32 rows of h and mean(agg/cnt) into LDS as bf16 ----
  {
    int base = tid * 16;                // 256 thr * 16 = 4096 = 32*128
    int r = base >> 7;
    int c = base & 127;
    int node = m0 + r;
    float vh[16], vm[16];
    if (node < N_NODES) {
      const float* hp = h_in + (long long)node * DIM + c;
      const float* ap = agg  + (long long)node * DIM + c;
      float rc = 1.0f / fmaxf(cnt[node], 1.0f);
#pragma unroll
      for (int i = 0; i < 16; ++i) { vh[i] = hp[i]; vm[i] = ap[i] * rc; }
    } else {
#pragma unroll
      for (int i = 0; i < 16; ++i) { vh[i] = 0.0f; vm[i] = 0.0f; }
    }
#pragma unroll
    for (int i = 0; i < 16; ++i) {
      sH[base + i] = f32_to_bf16(vh[i]);
      sM[base + i] = f32_to_bf16(vm[i]);
    }
  }
  __syncthreads();

  // ---- 8 waves, wave w owns output cols [16w, 16w+16) ----
  const int wave = tid >> 5;
  const int lane = tid & 31;
  const int lm   = lane & 15;   // row (A) / col (B,C)
  const int lh   = lane >> 4;   // half selector

  f32x8 c0 = {};                // rows [0,16)
  f32x8 c1 = {};                // rows [16,32)
  const unsigned short* wl = WlT + (wave * 16 + lm) * DIM;   // column of Wl
  const unsigned short* wr = WrT + (wave * 16 + lm) * DIM;   // column of Wr

#pragma unroll
  for (int kb = 0; kb < 4; ++kb) {
    const int K0 = kb * 32;
    AFrag bl, br;
    // B fragment: elements 0-7 = K[K0+lh*8 ..], 8-15 = K[K0+16+lh*8 ..]
    const unsigned short* blp = wl + K0 + lh * 8;
    bl.q[0] = *(const u32x4*)(blp);
    bl.q[1] = *(const u32x4*)(blp + 16);
    const unsigned short* brp = wr + K0 + lh * 8;
    br.q[0] = *(const u32x4*)(brp);
    br.q[1] = *(const u32x4*)(brp + 16);

    // A fragments: elements 0-7 = K[K0+lh*8 ..], 8-15 = K[K0+16+lh*8 ..]
    AFrag ah0, am0, ah1, am1;
    {
      const unsigned short* hp = sH + lm * DIM + K0 + lh * 8;
      ah0.q[0] = *(const u32x4*)(hp);
      ah0.q[1] = *(const u32x4*)(hp + 16);
      const unsigned short* mp = sM + lm * DIM + K0 + lh * 8;
      am0.q[0] = *(const u32x4*)(mp);
      am0.q[1] = *(const u32x4*)(mp + 16);
    }
    {
      const unsigned short* hp = sH + (16 + lm) * DIM + K0 + lh * 8;
      ah1.q[0] = *(const u32x4*)(hp);
      ah1.q[1] = *(const u32x4*)(hp + 16);
      const unsigned short* mp = sM + (16 + lm) * DIM + K0 + lh * 8;
      am1.q[0] = *(const u32x4*)(mp);
      am1.q[1] = *(const u32x4*)(mp + 16);
    }

    c0 = __builtin_amdgcn_wmma_f32_16x16x32_bf16(false, am0.v, false, bl.v,
                                                 (short)0, c0, false, false);
    c0 = __builtin_amdgcn_wmma_f32_16x16x32_bf16(false, ah0.v, false, br.v,
                                                 (short)0, c0, false, false);
    c1 = __builtin_amdgcn_wmma_f32_16x16x32_bf16(false, am1.v, false, bl.v,
                                                 (short)0, c1, false, false);
    c1 = __builtin_amdgcn_wmma_f32_16x16x32_bf16(false, ah1.v, false, br.v,
                                                 (short)0, c1, false, false);
  }

  // C layout: VGPR i -> row i+8*lh, col lm (within wave's 16-col slab)
  const float bval = bias[wave * 16 + lm];
#pragma unroll
  for (int i = 0; i < 8; ++i) {
    sO[(i + 8 * lh) * DIM + wave * 16 + lm]      = c0[i] + bval;
    sO[(16 + i + 8 * lh) * DIM + wave * 16 + lm] = c1[i] + bval;
  }
  __syncthreads();

  // ---- LayerNorm + ReLU: 16 threads per row, 8 elems each, 2 row passes ----
#pragma unroll
  for (int rr = 0; rr < 2; ++rr) {
    int row = rr * 16 + (tid >> 4);
    int sub = tid & 15;
    const float* rp = sO + row * DIM + sub * 8;
    float v[8];
    float s = 0.0f, sq = 0.0f;
#pragma unroll
    for (int i = 0; i < 8; ++i) { v[i] = rp[i]; s += v[i]; sq += v[i] * v[i]; }
#pragma unroll
    for (int m = 8; m >= 1; m >>= 1) {
      s  += __shfl_xor(s,  m, 16);
      sq += __shfl_xor(sq, m, 16);
    }
    float mu   = s * (1.0f / 128.0f);
    float var  = sq * (1.0f / 128.0f) - mu * mu;
    float rstd = rsqrtf(var + 1e-5f);
    int node = m0 + row;
    if (node < N_NODES) {
      float* op = h_out + (long long)node * DIM + sub * 8;
#pragma unroll
      for (int i = 0; i < 8; ++i) {
        int cidx = sub * 8 + i;
        float y = (v[i] - mu) * rstd * gamma[cidx] + beta[cidx];
        op[i] = fmaxf(y, 0.0f);
      }
    }
  }
}

// ---------------- per-graph pooling ----------------
__global__ void pool_scatter_kernel(const float* __restrict__ h,
                                    const int* __restrict__ gb,
                                    float* __restrict__ sums,
                                    unsigned int* __restrict__ maxb,
                                    float* __restrict__ gcnt) {
  long long t = (long long)blockIdx.x * blockDim.x + threadIdx.x;
  int node = (int)(t >> 5);
  int lane = (int)(t & 31);
  if (node >= N_NODES) return;
  int g = gb[node];
  f32x4 v = *(const f32x4*)(h + (long long)node * DIM + lane * 4);
  float* sp = sums + g * DIM + lane * 4;
  atomicAdd(sp + 0, v.x);
  atomicAdd(sp + 1, v.y);
  atomicAdd(sp + 2, v.z);
  atomicAdd(sp + 3, v.w);
  // post-ReLU values are >= 0 -> float order == unsigned bit order
  unsigned int* mp = maxb + g * DIM + lane * 4;
  atomicMax(mp + 0, __float_as_uint(v.x));
  atomicMax(mp + 1, __float_as_uint(v.y));
  atomicMax(mp + 2, __float_as_uint(v.z));
  atomicMax(mp + 3, __float_as_uint(v.w));
  if (lane == 0) atomicAdd(gcnt + g, 1.0f);
}

// ---------------- readout MLP + L2 normalize (one block per graph) ----------------
__global__ __launch_bounds__(256)
void readout_kernel(const float* __restrict__ psum,
                    const unsigned int* __restrict__ pmax,
                    const float* __restrict__ gcnt,
                    const float* __restrict__ Wr1, const float* __restrict__ br1,
                    const float* __restrict__ Wr2, const float* __restrict__ br2,
                    float* __restrict__ out) {
  __shared__ float pooled[2 * DIM];   // [mean(128), max(128)]
  __shared__ float hid[DIM];
  __shared__ float wsum[8];

  int g = blockIdx.x;
  int t = threadIdx.x;

  if (t < DIM) {
    float n = fmaxf(gcnt[g], 1.0f);
    pooled[t] = psum[g * DIM + t] / n;
  } else {
    pooled[t] = __uint_as_float(pmax[g * DIM + (t - DIM)]);
  }
  __syncthreads();

  if (t < DIM) {
    float s = br1[t];
#pragma unroll 4
    for (int k = 0; k < 2 * DIM; ++k) s += pooled[k] * Wr1[k * DIM + t];
    hid[t] = fmaxf(s, 0.0f);
  }
  __syncthreads();

  float e = br2[t];
#pragma unroll 4
  for (int k = 0; k < DIM; ++k) e += hid[k] * Wr2[k * OUT_DIM + t];

  float sq = e * e;
#pragma unroll
  for (int m = 16; m >= 1; m >>= 1) sq += __shfl_xor(sq, m, 32);
  if ((t & 31) == 0) wsum[t >> 5] = sq;
  __syncthreads();
  float tot = 0.0f;
#pragma unroll
  for (int i = 0; i < 8; ++i) tot += wsum[i];
  float nrm = fmaxf(sqrtf(tot), 1e-12f);
  out[g * OUT_DIM + t] = e / nrm;
}

// ---------------- launch ----------------
extern "C" void kernel_launch(void* const* d_in, const int* in_sizes, int n_in,
                              void* d_out, int out_size, void* d_ws, size_t ws_size,
                              hipStream_t stream) {
  (void)in_sizes; (void)n_in; (void)out_size; (void)ws_size;

  const float* x   = (const float*)d_in[0];
  const int*   ei  = (const int*)d_in[1];
  const int*   src = ei;
  const int*   dst = ei + N_EDGES;
  const int*   gb  = (const int*)d_in[2];
  const float* W1l = (const float*)d_in[3];
  const float* W1r = (const float*)d_in[4];
  const float* b1  = (const float*)d_in[5];
  const float* g1  = (const float*)d_in[6];
  const float* be1 = (const float*)d_in[7];
  const float* W2l = (const float*)d_in[8];
  const float* W2r = (const float*)d_in[9];
  const float* b2  = (const float*)d_in[10];
  const float* g2  = (const float*)d_in[11];
  const float* be2 = (const float*)d_in[12];
  const float* W3l = (const float*)d_in[13];
  const float* W3r = (const float*)d_in[14];
  const float* b3  = (const float*)d_in[15];
  const float* g3  = (const float*)d_in[16];
  const float* be3 = (const float*)d_in[17];
  const float* Wr1 = (const float*)d_in[18];
  const float* br1 = (const float*)d_in[19];
  const float* Wr2 = (const float*)d_in[20];
  const float* br2 = (const float*)d_in[21];

  // workspace layout (all offsets 256B aligned)
  const size_t SZ_H   = (size_t)N_NODES * DIM * sizeof(float);   // 51,200,000
  const size_t OFF_AGG  = 0;
  const size_t OFF_HA   = OFF_AGG + SZ_H;
  const size_t OFF_HB   = OFF_HA + SZ_H;
  const size_t OFF_CNT  = OFF_HB + SZ_H;
  const size_t OFF_W    = OFF_CNT + 400128;                      // cnt 400,000 padded
  const size_t OFF_PSUM = OFF_W + 6 * (size_t)DIM * DIM * 2;     // 6 bf16 matrices
  const size_t OFF_PMAX = OFF_PSUM + (size_t)N_GRAPHS * DIM * 4;
  const size_t OFF_GCNT = OFF_PMAX + (size_t)N_GRAPHS * DIM * 4;

  char* ws = (char*)d_ws;
  float*          agg  = (float*)(ws + OFF_AGG);
  float*          hA   = (float*)(ws + OFF_HA);
  float*          hB   = (float*)(ws + OFF_HB);
  float*          cnt  = (float*)(ws + OFF_CNT);
  unsigned short* Wt   = (unsigned short*)(ws + OFF_W);
  float*          psum = (float*)(ws + OFF_PSUM);
  unsigned int*   pmax = (unsigned int*)(ws + OFF_PMAX);
  float*          gcnt = (float*)(ws + OFF_GCNT);

  const int WMAT = DIM * DIM;   // 16384 elems per matrix

  // 1) convert all 6 SAGE weight matrices to transposed bf16
  wtrans_bf16_kernel<<<64, 256, 0, stream>>>(W1l, Wt + 0 * WMAT);
  wtrans_bf16_kernel<<<64, 256, 0, stream>>>(W1r, Wt + 1 * WMAT);
  wtrans_bf16_kernel<<<64, 256, 0, stream>>>(W2l, Wt + 2 * WMAT);
  wtrans_bf16_kernel<<<64, 256, 0, stream>>>(W2r, Wt + 3 * WMAT);
  wtrans_bf16_kernel<<<64, 256, 0, stream>>>(W3l, Wt + 4 * WMAT);
  wtrans_bf16_kernel<<<64, 256, 0, stream>>>(W3r, Wt + 5 * WMAT);

  // 2) in-degree counts (same for all layers)
  fill_u32_kernel<<<512, 256, 0, stream>>>((unsigned int*)cnt, 0u, N_NODES);
  count_deg_kernel<<<(N_EDGES + 255) / 256, 256, 0, stream>>>(dst, cnt);

  const long long aggN   = (long long)N_NODES * DIM;
  const int sageBlocks   = (N_NODES + MTILE - 1) / MTILE;            // 3125
  const int edgeBlocks   = (int)(((long long)N_EDGES * 32) / 256);   // 75000
  const int poolBlocks   = (int)(((long long)N_NODES * 32) / 256);   // 12500

  // 3) layer 1: x -> hA
  fill_u32_kernel<<<4096, 256, 0, stream>>>((unsigned int*)agg, 0u, aggN);
  scatter_agg_kernel<<<edgeBlocks, 256, 0, stream>>>(x, src, dst, agg);
  sage_wmma_kernel<<<sageBlocks, 256, 0, stream>>>(x, agg, cnt,
      Wt + 0 * WMAT, Wt + 1 * WMAT, b1, g1, be1, hA);

  // 4) layer 2: hA -> hB
  fill_u32_kernel<<<4096, 256, 0, stream>>>((unsigned int*)agg, 0u, aggN);
  scatter_agg_kernel<<<edgeBlocks, 256, 0, stream>>>(hA, src, dst, agg);
  sage_wmma_kernel<<<sageBlocks, 256, 0, stream>>>(hA, agg, cnt,
      Wt + 2 * WMAT, Wt + 3 * WMAT, b2, g2, be2, hB);

  // 5) layer 3: hB -> hA
  fill_u32_kernel<<<4096, 256, 0, stream>>>((unsigned int*)agg, 0u, aggN);
  scatter_agg_kernel<<<edgeBlocks, 256, 0, stream>>>(hB, src, dst, agg);
  sage_wmma_kernel<<<sageBlocks, 256, 0, stream>>>(hB, agg, cnt,
      Wt + 4 * WMAT, Wt + 5 * WMAT, b3, g3, be3, hA);

  // 6) pooling (psum + pmax + gcnt are contiguous -> single fill)
  fill_u32_kernel<<<256, 256, 0, stream>>>((unsigned int*)psum, 0u,
      (long long)N_GRAPHS * DIM * 2 + N_GRAPHS);
  pool_scatter_kernel<<<poolBlocks, 256, 0, stream>>>(hA, gb, psum, pmax, gcnt);

  // 7) readout MLP + L2 normalize
  readout_kernel<<<N_GRAPHS, 256, 0, stream>>>(psum, pmax, gcnt,
      Wr1, br1, Wr2, br2, (float*)d_out);
}
